// MyLSTMCell_17102559772818
// MI455X (gfx1250) — matrix operand order repaired and checked by reading, not compiled, not run on hardware
//
#include <hip/hip_runtime.h>
#include <hip/hip_bf16.h>

// ---------------------------------------------------------------------------
// Fused LSTM cell for MI455X (gfx1250, wave32, WMMA).
//   gates = [prev_h | input] @ [W_i;W_f;W_o;W_g]^T  (16384x1024 @ 1024x2048)
// v2: block = 8 waves sharing one 16-column tile; the 4 gate B-tiles are
// staged through LDS (double-buffered 128-half K panels) so B global traffic
// is amortized 8x across the block. A streams straight from L2 (whole f16 X
// + W = 36MB fits in the 192MB L2). WMMA f32 accumulate, f16 inputs.
// ---------------------------------------------------------------------------

#define BATCH 16384
#define HID   512
#define KDIM  1024            // INPUT + HIDDEN
#define KC    128             // K panel staged in LDS per barrier period
#define NPANEL (KDIM / KC)    // 8
#define LDS_STRIDE (KC + 8)   // halves; +16B pad -> rows hit distinct banks

typedef __attribute__((ext_vector_type(16))) _Float16 v16h;
typedef __attribute__((ext_vector_type(4)))  _Float16 v4h;
typedef __attribute__((ext_vector_type(8)))  float    v8f;

// ---- f16 tile loaders: A and B share contiguous k = 16*(L>>4)+s order -------
// A 16x32: lane L = row (L&15); contiguous 16 halves starting at
//          k-offset (L>>4)*16  -> two adjacent 16B chunks.
__device__ __forceinline__ v16h loadA(const _Float16* p) {
    v16h r;
    ((float4*)&r)[0] = *(const float4*)(p);       // halves [0..8)
    ((float4*)&r)[1] = *(const float4*)(p + 8);   // halves [8..16)
    return r;
}
// B 32x16: lane L = column (L&15) (== a row of W since B = W^T);
//          contiguous 16 halves starting at k-offset (L>>4)*16.
__device__ __forceinline__ v16h loadB_lds(const _Float16* p) {
    v16h r;
    ((float4*)&r)[0] = *(const float4*)(p);       // ds_load_b128
    ((float4*)&r)[1] = *(const float4*)(p + 8);   // ds_load_b128
    return r;
}

__device__ __forceinline__ float sigmoid_f(float x) {
    return 1.0f / (1.0f + __expf(-x));
}

// ---- pack X = concat(prev_h, input_) per row, fp32 -> f16 -----------------
__global__ __launch_bounds__(256) void pack_x_kernel(
    const float* __restrict__ input_, const float* __restrict__ prev_h,
    _Float16* __restrict__ Xh)
{
    int idx = blockIdx.x * 256 + threadIdx.x;   // [0, BATCH*KDIM/4)
    int row = idx >> 8;                         // KDIM/4 = 256 quads per row
    int k   = (idx & 255) * 4;
    const float* src = (k < HID) ? (prev_h + (size_t)row * HID + k)
                                 : (input_ + (size_t)row * HID + (k - HID));
    float4 v = *(const float4*)src;
    v4h o = { (_Float16)v.x, (_Float16)v.y, (_Float16)v.z, (_Float16)v.w };
    *(v4h*)(Xh + (size_t)row * KDIM + k) = o;
}

// ---- pack the 4 gate weight matrices fp32 -> f16 (order: i, f, o, g) ------
__global__ __launch_bounds__(256) void pack_w_kernel(
    const float* __restrict__ Wi, const float* __restrict__ Wf,
    const float* __restrict__ Wo, const float* __restrict__ Wg,
    _Float16* __restrict__ Wh)
{
    int g = blockIdx.y;
    const float* src = (g == 0) ? Wi : (g == 1) ? Wf : (g == 2) ? Wo : Wg;
    size_t idx = (size_t)blockIdx.x * 256 + threadIdx.x;   // [0, HID*KDIM/4)
    float4 v = *(const float4*)(src + idx * 4);
    v4h o = { (_Float16)v.x, (_Float16)v.y, (_Float16)v.z, (_Float16)v.w };
    *(v4h*)(Wh + (size_t)g * HID * KDIM + idx * 4) = o;
}

// ---- fused 4-gate WMMA GEMM + LSTM elementwise ----------------------------
// Block = 256 threads = 8 waves, all on the same 16-col tile; wave w owns
// rows [row_blk*512 + w*64, +64). B tiles staged in LDS, double-buffered.
// Grid: 32 row-blocks * 32 col-tiles = 1024 blocks.
__global__ __launch_bounds__(256) void lstm_wmma_kernel(
    const _Float16* __restrict__ Xh, const _Float16* __restrict__ Wh,
    const float* __restrict__ b_i, const float* __restrict__ b_f,
    const float* __restrict__ b_o, const float* __restrict__ b_g,
    const float* __restrict__ prev_c,
    float* __restrict__ out_h, float* __restrict__ out_c)
{
    // [buf][gate*16 + col][k]   (rows padded to kill bank conflicts)
    __shared__ _Float16 Bs[2][64][LDS_STRIDE];

    const int tid      = threadIdx.x;
    const int lane     = tid & 31;
    const int wblk     = tid >> 5;            // wave in block: 0..7
    const int n_tile   = blockIdx.x & 31;     // 512/16 = 32 column tiles
    const int row_blk  = blockIdx.x >> 5;     // 16384/512 = 32 row blocks
    const int row_base = row_blk * 512 + wblk * 64;
    const int n_base   = n_tile * 16;
    const int lm = lane & 15;
    const int hi = lane >> 4;

    // A: lane holds batch row (row_base + t*16 + lm)
    const _Float16* aptr = Xh + (size_t)(row_base + lm) * KDIM + hi * 16;

    // B staging: thread t handles W-row column (t>>4), 16B chunk (t&15),
    // one chunk per gate -> 4 float4 loads fill a full 4x16xKC panel.
    const int scol   = tid >> 4;      // 0..15
    const int schunk = tid & 15;      // 0..15
    const _Float16* wsrc[4];
#pragma unroll
    for (int g = 0; g < 4; ++g)
        wsrc[g] = Wh + ((size_t)g * HID + n_base + scol) * KDIM + schunk * 8;
    _Float16* sdst0 = &Bs[0][0][0] + (size_t)scol * LDS_STRIDE + schunk * 8;
    _Float16* sdst1 = &Bs[1][0][0] + (size_t)scol * LDS_STRIDE + schunk * 8;
    const size_t gate_lds = (size_t)16 * LDS_STRIDE;

    // prologue: fill buffer 0 with panel kb=0
#pragma unroll
    for (int g = 0; g < 4; ++g)
        *(float4*)(sdst0 + g * gate_lds) = *(const float4*)(wsrc[g]);
    __syncthreads();

    v8f acc[4][4] = {};   // [gate i,f,o,g][row subtile]
    float4 stg[4];

    for (int kb = 0; kb < NPANEL; ++kb) {
        const int kbase = kb * KC;
        const _Float16* bs_cur = (kb & 1) ? &Bs[1][0][0] : &Bs[0][0][0];
        _Float16*       bs_nxt = (kb & 1) ? sdst0 : sdst1;

        if (kb + 1 < NPANEL) {      // prefetch next panel into registers
#pragma unroll
            for (int g = 0; g < 4; ++g)
                stg[g] = *(const float4*)(wsrc[g] + kbase + KC);
        }

#pragma unroll
        for (int k0 = 0; k0 < KC; k0 += 32) {
            const _Float16* bl =
                bs_cur + (size_t)lm * LDS_STRIDE + k0 + hi * 16;
            v16h B0 = loadB_lds(bl + 0 * gate_lds);
            v16h B1 = loadB_lds(bl + 1 * gate_lds);
            v16h B2 = loadB_lds(bl + 2 * gate_lds);
            v16h B3 = loadB_lds(bl + 3 * gate_lds);
#pragma unroll
            for (int t = 0; t < 4; ++t) {
                v16h A = loadA(aptr + (size_t)t * 16 * KDIM + kbase + k0);
                acc[0][t] = __builtin_amdgcn_wmma_f32_16x16x32_f16(
                    false, A, false, B0, (short)0, acc[0][t], false, false);
                acc[1][t] = __builtin_amdgcn_wmma_f32_16x16x32_f16(
                    false, A, false, B1, (short)0, acc[1][t], false, false);
                acc[2][t] = __builtin_amdgcn_wmma_f32_16x16x32_f16(
                    false, A, false, B2, (short)0, acc[2][t], false, false);
                acc[3][t] = __builtin_amdgcn_wmma_f32_16x16x32_f16(
                    false, A, false, B3, (short)0, acc[3][t], false, false);
            }
        }

        if (kb + 1 < NPANEL) {
            __syncthreads();        // everyone done reading the other buffer
#pragma unroll
            for (int g = 0; g < 4; ++g)
                *(float4*)(bs_nxt + g * gate_lds) = stg[g];
            __syncthreads();        // panel visible before next compute
        }
    }

    // Epilogue. C/D layout: VGPR j <-> M = j + 8*(lane>=16), lane&15 <-> N.
    const int col = n_base + lm;
    const float bI = b_i[col];
    const float bF = b_f[col];
    const float bO = b_o[col];
    const float bG = b_g[col];

#pragma unroll
    for (int t = 0; t < 4; ++t) {
#pragma unroll
        for (int j = 0; j < 8; ++j) {
            const int m = row_base + t * 16 + j + hi * 8;
            const size_t off = (size_t)m * HID + col;
            float gi = sigmoid_f(acc[0][t][j] + bI);
            float gf = sigmoid_f(acc[1][t][j] + bF);
            float go = sigmoid_f(acc[2][t][j] + bO);
            float gg = tanhf(acc[3][t][j] + bG);
            float c  = gf * prev_c[off] + gi * gg;
            float h  = tanhf(c) * go;
            out_h[off] = h;
            out_c[off] = c;
        }
    }
}

extern "C" void kernel_launch(void* const* d_in, const int* in_sizes, int n_in,
                              void* d_out, int out_size, void* d_ws, size_t ws_size,
                              hipStream_t stream) {
    const float* input_ = (const float*)d_in[0];
    const float* prev_h = (const float*)d_in[1];
    const float* prev_c = (const float*)d_in[2];
    const float* W_i    = (const float*)d_in[3];
    const float* b_i    = (const float*)d_in[4];
    const float* W_f    = (const float*)d_in[5];
    const float* b_f    = (const float*)d_in[6];
    const float* W_g    = (const float*)d_in[7];
    const float* b_g    = (const float*)d_in[8];
    const float* W_o    = (const float*)d_in[9];
    const float* b_o    = (const float*)d_in[10];

    // Workspace layout (f16): Xh[16384*1024] then Wh[4*512*1024] (i,f,o,g).
    _Float16* Xh = (_Float16*)d_ws;
    _Float16* Wh = Xh + (size_t)BATCH * KDIM;

    float* out_h = (float*)d_out;
    float* out_c = out_h + (size_t)BATCH * HID;

    // 1) pack X (16384*1024 elems, 4 per thread)
    pack_x_kernel<<<(BATCH * KDIM / 4) / 256, 256, 0, stream>>>(input_, prev_h, Xh);
    // 2) pack W (gate order i, f, o, g)
    pack_w_kernel<<<dim3((HID * KDIM / 4) / 256, 4), 256, 0, stream>>>(
        W_i, W_f, W_o, W_g, Wh);
    // 3) fused WMMA GEMM + LSTM pointwise: 8192 waves -> 1024 blocks
    lstm_wmma_kernel<<<1024, 256, 0, stream>>>(
        Xh, Wh, b_i, b_f, b_o, b_g, prev_c, out_h, out_c);
}
